// BGGNMixtureBernoulli_82686710383410
// MI455X (gfx1250) — matrix operand order repaired and checked by reading, not compiled, not run on hardware
//
#include <hip/hip_runtime.h>
#include <hip/hip_bf16.h>

// ---------------- problem constants (match reference) ----------------
static constexpr int HD  = 256;   // hidden / embedding dim
static constexpr int EFD = 128;   // edge feature dim
static constexpr int AHD = 128;   // attention hidden dim
static constexpr int KM  = 20;    // mixture components
static constexpr int KMP = 32;    // padded to WMMA tile multiple
static constexpr int LN  = 2;     // GNN layers
static constexpr int LDE = 392;   // LDS row stride (halves) for edge_input rows, 16B aligned
static constexpr int LDH = 264;   // LDS row stride (halves) for 256-col tiles, 16B aligned

typedef _Float16 v16h __attribute__((ext_vector_type(16)));
typedef _Float16 v8h  __attribute__((ext_vector_type(8)));
typedef float    v8f  __attribute__((ext_vector_type(8)));

// ---------------- WMMA helpers (gfx1250: V_WMMA_F32_16X16X32_F16) ----------------
__device__ __forceinline__ v8f wmma16(v16h a, v16h b, v8f c) {
  // (neg_a, A, neg_b, B, c_mod, C, reuse_a, reuse_b)
  return __builtin_amdgcn_wmma_f32_16x16x32_f16(false, a, false, b, (short)0, c, false, false);
}

// A fragment 16x32 f16 from LDS (row-major, stride lda halves).
// ISA layout: lanes 0-15 hold row m=lane, K {0..7,16..23}; lanes 16-31 row m=lane-16, K {8..15,24..31}.
__device__ __forceinline__ v16h frag_a_lds(const _Float16* base, int lda, int kt, int lane) {
  const int m   = lane & 15;
  const int klo = (lane & 16) ? 8 : 0;
  const _Float16* p = base + m * lda + kt * 32 + klo;
  v16h r;
  ((v8h*)&r)[0] = *(const v8h*)(p);        // K klo .. klo+7
  ((v8h*)&r)[1] = *(const v8h*)(p + 16);   // K klo+16 .. klo+23
  return r;
}

// B fragment 32x16 f16 from global weight W stored row-major [out, kdim] (B[k][n] = W[n][k]).
// ISA layout: lane n=lane&15 is the column; lanes 0-15 hold K {0..7,16..23}, lanes 16-31 hold K {8..15,24..31}.
__device__ __forceinline__ v16h frag_b_glb(const _Float16* __restrict__ W, int kdim,
                                           int n0, int kt, int lane) {
  const int n = n0 + (lane & 15);
  const int k = kt * 32 + ((lane & 16) ? 8 : 0);
  const _Float16* p = W + (size_t)n * kdim + k;
  v16h r;
  ((v8h*)&r)[0] = *(const v8h*)(p);
  ((v8h*)&r)[1] = *(const v8h*)(p + 16);
  return r;
}

__device__ __forceinline__ float sigm(float x) { return 1.0f / (1.0f + __expf(-x)); }

// ---------------- tiny utility kernels ----------------
__global__ void zero_f32_kernel(float* p, size_t n) {
  size_t i = (size_t)blockIdx.x * blockDim.x + threadIdx.x;
  if (i < n) p[i] = 0.0f;
}

__global__ void f32_to_f16_kernel(const float* __restrict__ src, _Float16* __restrict__ dst, size_t n) {
  size_t i = (size_t)blockIdx.x * blockDim.x + threadIdx.x;
  if (i < n) dst[i] = (_Float16)src[i];
}

// pad [KM,HD] -> [KMP,HD] f16 (extra rows zero)
__global__ void pad_w3_kernel(const float* __restrict__ src, _Float16* __restrict__ dst) {
  int i = blockIdx.x * blockDim.x + threadIdx.x;
  if (i < KMP * HD) {
    int r = i / HD, c = i % HD;
    dst[i] = (_Float16)((r < KM) ? src[r * HD + c] : 0.0f);
  }
}

__global__ void sum_pos_kernel(const float* __restrict__ pos, float* __restrict__ sums, int n) {
  __shared__ float red[256];
  int tid = threadIdx.x;
  float acc = 0.0f;
  for (size_t i = (size_t)blockIdx.x * blockDim.x + tid; i < (size_t)n;
       i += (size_t)gridDim.x * blockDim.x)
    acc += pos[i];
  red[tid] = acc; __syncthreads();
  for (int s = 128; s > 0; s >>= 1) { if (tid < s) red[tid] += red[tid + s]; __syncthreads(); }
  if (tid == 0) atomicAdd(&sums[0], red[0]);
}

// state init (items_emb) + BPR loss per node
__global__ void node_init_kernel(const int* __restrict__ user_idx,
                                 const int* __restrict__ node_idx_feat,
                                 const float* __restrict__ pos_idx,
                                 const float* __restrict__ users_feature,
                                 const float* __restrict__ items_feature,
                                 const float* __restrict__ sums,
                                 float* __restrict__ state,
                                 float* __restrict__ bpr, int n) {
  int i = blockIdx.x * blockDim.x + threadIdx.x;
  if (i >= n) return;
  const float* u  = users_feature + (size_t)user_idx[i] * HD;
  const float* it = items_feature + (size_t)node_idx_feat[i] * HD;
  float* st = state + (size_t)i * HD;
  float dot = 0.0f;
  for (int k = 0; k < HD; k += 4) {
    float4 uv = *(const float4*)(u + k);
    float4 iv = *(const float4*)(it + k);
    dot += uv.x * iv.x + uv.y * iv.y + uv.z * iv.z + uv.w * iv.w;
    *(float4*)(st + k) = iv;
  }
  float p  = pos_idx[i];
  float sp = sums[0];
  float sn = (float)n - sp;
  // softplus(neg_score - pos_score); user.pos = p*dot, user.neg = (1-p)*dot
  float x = dot * ((1.0f - p) / sn - p / sp);
  bpr[i] = (x > 20.0f) ? x : log1pf(__expf(x));
}

// per-layer prep: optional relu on f32 state, emit f16 copy, zero agg
__global__ void prep_state_kernel(float* __restrict__ state, _Float16* __restrict__ state_h,
                                  float* __restrict__ agg, int do_relu, size_t n) {
  size_t i = (size_t)blockIdx.x * blockDim.x + threadIdx.x;
  if (i < n) {
    float s = state[i];
    if (do_relu) { s = s > 0.0f ? s : 0.0f; state[i] = s; }
    state_h[i] = (_Float16)s;
    agg[i] = 0.0f;
  }
}

// ---------------- edge message + attention kernel (fused, WMMA) ----------------
// block = 256 threads (8 waves), handles 32 edges (2 M-tiles, B fragments reused 2x).
__global__ __launch_bounds__(256)
void edge_msg_kernel(const _Float16* __restrict__ state_h,
                     const int* __restrict__ edges,
                     const int* __restrict__ att_idx,
                     const float* __restrict__ att_noise,
                     const _Float16* __restrict__ W1, const float* __restrict__ b1,
                     const _Float16* __restrict__ W2, const float* __restrict__ b2,
                     const _Float16* __restrict__ A1, const float* __restrict__ ab1,
                     const _Float16* __restrict__ A2, const float* __restrict__ ab2,
                     float* __restrict__ agg) {
  __shared__ __align__(16) _Float16 s_ein[32][LDE];  // edge_input [32, 384]
  __shared__ __align__(16) _Float16 s_hid[32][LDH];  // hidden (msg 256, then att 128)
  __shared__ __align__(16) _Float16 s_msg[32][LDH];  // msg output [32, 256]
  __shared__ int s_src[32], s_dst[32], s_flag[32];

  const int e0 = blockIdx.x * 32;
  const int tid = threadIdx.x, lane = tid & 31, wave = tid >> 5;

  if (tid < 32) {
    int e = e0 + tid;
    int s = edges[2 * e], d = edges[2 * e + 1];
    s_src[tid] = s; s_dst[tid] = d;
    s_flag[tid] = (att_idx[s] > 0) || (att_idx[d] > 0);
  }
  __syncthreads();

  // stage edge_input: diff (cols 0..255) + masked noise (cols 256..383)
  {
    const int m = tid >> 3;             // 0..31
    const int q = tid & 7;              // 0..7
    const int c = q * 32;               // 0..224
    const _Float16* ps = state_h + (size_t)s_src[m] * HD + c;
    const _Float16* pd = state_h + (size_t)s_dst[m] * HD + c;
    #pragma unroll
    for (int j = 0; j < 4; ++j)
      *(v8h*)&s_ein[m][c + 8 * j] = *(const v8h*)(ps + 8 * j) - *(const v8h*)(pd + 8 * j);
    const int fc = q * 16;              // 0..112
    const float fl = s_flag[m] ? 1.0f : 0.0f;
    const float* pn = att_noise + (size_t)(e0 + m) * EFD + fc;
    #pragma unroll
    for (int j = 0; j < 16; ++j) s_ein[m][HD + fc + j] = (_Float16)(fl * pn[j]);
  }
  __syncthreads();

  // msg hidden: relu(edge_input @ W1^T + b1)   [32,256], K=384
  #pragma unroll
  for (int j = 0; j < 2; ++j) {
    const int n0 = (wave * 2 + j) * 16;
    v8f acc0 = {}, acc1 = {};
    #pragma unroll
    for (int kt = 0; kt < (HD + EFD) / 32; ++kt) {
      v16h b = frag_b_glb(W1, HD + EFD, n0, kt, lane);
      acc0 = wmma16(frag_a_lds(&s_ein[0][0],  LDE, kt, lane), b, acc0);
      acc1 = wmma16(frag_a_lds(&s_ein[16][0], LDE, kt, lane), b, acc1);
    }
    const int n = n0 + (lane & 15);
    const float bias = b1[n];
    const int mb = (lane & 16) ? 8 : 0;
    #pragma unroll
    for (int r = 0; r < 8; ++r) {
      float v0 = acc0[r] + bias, v1 = acc1[r] + bias;
      s_hid[mb + r][n]      = (_Float16)(v0 > 0.0f ? v0 : 0.0f);
      s_hid[16 + mb + r][n] = (_Float16)(v1 > 0.0f ? v1 : 0.0f);
    }
  }
  __syncthreads();

  // msg out: hidden @ W2^T + b2   [32,256], K=256
  #pragma unroll
  for (int j = 0; j < 2; ++j) {
    const int n0 = (wave * 2 + j) * 16;
    v8f acc0 = {}, acc1 = {};
    #pragma unroll
    for (int kt = 0; kt < HD / 32; ++kt) {
      v16h b = frag_b_glb(W2, HD, n0, kt, lane);
      acc0 = wmma16(frag_a_lds(&s_hid[0][0],  LDH, kt, lane), b, acc0);
      acc1 = wmma16(frag_a_lds(&s_hid[16][0], LDH, kt, lane), b, acc1);
    }
    const int n = n0 + (lane & 15);
    const float bias = b2[n];
    const int mb = (lane & 16) ? 8 : 0;
    #pragma unroll
    for (int r = 0; r < 8; ++r) {
      s_msg[mb + r][n]      = (_Float16)(acc0[r] + bias);
      s_msg[16 + mb + r][n] = (_Float16)(acc1[r] + bias);
    }
  }
  __syncthreads();

  // att hidden: relu(edge_input @ A1^T + ab1)   [32,128], K=384 (reuse s_hid)
  {
    const int n0 = wave * 16;                  // 8 waves cover 128 cols
    v8f acc0 = {}, acc1 = {};
    #pragma unroll
    for (int kt = 0; kt < (HD + EFD) / 32; ++kt) {
      v16h b = frag_b_glb(A1, HD + EFD, n0, kt, lane);
      acc0 = wmma16(frag_a_lds(&s_ein[0][0],  LDE, kt, lane), b, acc0);
      acc1 = wmma16(frag_a_lds(&s_ein[16][0], LDE, kt, lane), b, acc1);
    }
    const int n = n0 + (lane & 15);
    const float bias = ab1[n];
    const int mb = (lane & 16) ? 8 : 0;
    #pragma unroll
    for (int r = 0; r < 8; ++r) {
      float v0 = acc0[r] + bias, v1 = acc1[r] + bias;
      s_hid[mb + r][n]      = (_Float16)(v0 > 0.0f ? v0 : 0.0f);
      s_hid[16 + mb + r][n] = (_Float16)(v1 > 0.0f ? v1 : 0.0f);
    }
  }
  __syncthreads();

  // att out: sigmoid(atthid @ A2^T + ab2) * msg -> atomicAdd agg[dst]
  #pragma unroll
  for (int j = 0; j < 2; ++j) {
    const int n0 = (wave * 2 + j) * 16;
    v8f acc0 = {}, acc1 = {};
    #pragma unroll
    for (int kt = 0; kt < AHD / 32; ++kt) {
      v16h b = frag_b_glb(A2, AHD, n0, kt, lane);
      acc0 = wmma16(frag_a_lds(&s_hid[0][0],  LDH, kt, lane), b, acc0);
      acc1 = wmma16(frag_a_lds(&s_hid[16][0], LDH, kt, lane), b, acc1);
    }
    const int n = n0 + (lane & 15);
    const float bias = ab2[n];
    const int mb = (lane & 16) ? 8 : 0;
    #pragma unroll
    for (int r = 0; r < 8; ++r) {
      int m = mb + r;
      atomicAdd(&agg[(size_t)s_dst[m] * HD + n],
                sigm(acc0[r] + bias) * (float)s_msg[m][n]);
      m = 16 + mb + r;
      atomicAdd(&agg[(size_t)s_dst[m] * HD + n],
                sigm(acc1[r] + bias) * (float)s_msg[m][n]);
    }
  }
}

// ---------------- GRU cell kernel (fused gates, WMMA) ----------------
// block = 256 threads (8 waves), 32 nodes (2 M-tiles; B reused 2x). Tail block guarded.
__global__ __launch_bounds__(256)
void gru_kernel(const float* __restrict__ agg,
                const _Float16* __restrict__ state_h,
                const _Float16* __restrict__ Wih, const _Float16* __restrict__ Whh,
                const float* __restrict__ bih, const float* __restrict__ bhh,
                float* __restrict__ state, int Nn) {
  __shared__ __align__(16) _Float16 s_agg[32][LDH];
  __shared__ __align__(16) _Float16 s_h[32][LDH];
  const int i0 = blockIdx.x * 32;
  const int tid = threadIdx.x, lane = tid & 31, wave = tid >> 5;
  {
    const int m = tid >> 3, q = tid & 7;
    const int node = min(i0 + m, Nn - 1);     // clamp for tail block
    const int c = q * 32;
    const float* pa = agg + (size_t)node * HD + c;
    #pragma unroll
    for (int j = 0; j < 32; ++j) s_agg[m][c + j] = (_Float16)pa[j];
    const _Float16* ph = state_h + (size_t)node * HD + c;
    #pragma unroll
    for (int j = 0; j < 4; ++j) *(v8h*)&s_h[m][c + 8 * j] = *(const v8h*)(ph + 8 * j);
  }
  __syncthreads();
  #pragma unroll
  for (int j = 0; j < 2; ++j) {
    const int t = wave * 2 + j;                 // 16 col-tiles over H
    v8f gir0 = {}, giz0 = {}, gin0 = {}, ghr0 = {}, ghz0 = {}, ghn0 = {};
    v8f gir1 = {}, giz1 = {}, gin1 = {}, ghr1 = {}, ghz1 = {}, ghn1 = {};
    for (int kt = 0; kt < HD / 32; ++kt) {
      v16h aA0 = frag_a_lds(&s_agg[0][0],  LDH, kt, lane);
      v16h aA1 = frag_a_lds(&s_agg[16][0], LDH, kt, lane);
      v16h aH0 = frag_a_lds(&s_h[0][0],    LDH, kt, lane);
      v16h aH1 = frag_a_lds(&s_h[16][0],   LDH, kt, lane);
      v16h b;
      b = frag_b_glb(Wih, HD, t * 16,          kt, lane); gir0 = wmma16(aA0, b, gir0); gir1 = wmma16(aA1, b, gir1);
      b = frag_b_glb(Wih, HD, HD + t * 16,     kt, lane); giz0 = wmma16(aA0, b, giz0); giz1 = wmma16(aA1, b, giz1);
      b = frag_b_glb(Wih, HD, 2 * HD + t * 16, kt, lane); gin0 = wmma16(aA0, b, gin0); gin1 = wmma16(aA1, b, gin1);
      b = frag_b_glb(Whh, HD, t * 16,          kt, lane); ghr0 = wmma16(aH0, b, ghr0); ghr1 = wmma16(aH1, b, ghr1);
      b = frag_b_glb(Whh, HD, HD + t * 16,     kt, lane); ghz0 = wmma16(aH0, b, ghz0); ghz1 = wmma16(aH1, b, ghz1);
      b = frag_b_glb(Whh, HD, 2 * HD + t * 16, kt, lane); ghn0 = wmma16(aH0, b, ghn0); ghn1 = wmma16(aH1, b, ghn1);
    }
    const int n = t * 16 + (lane & 15);
    const float bir = bih[n], biz = bih[HD + n], bnn = bih[2 * HD + n];
    const float bhr = bhh[n], bhz = bhh[HD + n], bhn = bhh[2 * HD + n];
    const int mb = (lane & 16) ? 8 : 0;
    #pragma unroll
    for (int r = 0; r < 8; ++r) {
      int m = mb + r;
      if (i0 + m < Nn) {
        float rg = sigm((gir0[r] + bir) + (ghr0[r] + bhr));
        float zg = sigm((giz0[r] + biz) + (ghz0[r] + bhz));
        float nn = tanhf((gin0[r] + bnn) + rg * (ghn0[r] + bhn));
        float h  = (float)s_h[m][n];
        state[(size_t)(i0 + m) * HD + n] = (1.0f - zg) * nn + zg * h;
      }
      m = 16 + mb + r;
      if (i0 + m < Nn) {
        float rg = sigm((gir1[r] + bir) + (ghr1[r] + bhr));
        float zg = sigm((giz1[r] + biz) + (ghz1[r] + bhz));
        float nn = tanhf((gin1[r] + bnn) + rg * (ghn1[r] + bhn));
        float h  = (float)s_h[m][n];
        state[(size_t)(i0 + m) * HD + n] = (1.0f - zg) * nn + zg * h;
      }
    }
  }
}

// ---------------- output-head kernel (two fused 3-layer MLPs, WMMA) ----------------
// block = 256 threads (8 waves), 32 pairs (2 M-tiles; B reused 2x).
__global__ __launch_bounds__(256)
void head_kernel(const _Float16* __restrict__ state_h,
                 const int* __restrict__ gnn,
                 const _Float16* __restrict__ tw1, const float* __restrict__ tb1,
                 const _Float16* __restrict__ tw2, const float* __restrict__ tb2,
                 const _Float16* __restrict__ tw3, const float* __restrict__ tb3,
                 const _Float16* __restrict__ aw1, const float* __restrict__ ab1,
                 const _Float16* __restrict__ aw2, const float* __restrict__ ab2,
                 const _Float16* __restrict__ aw3, const float* __restrict__ ab3,
                 float* __restrict__ out0, float* __restrict__ out1) {
  __shared__ __align__(16) _Float16 s_diff[32][LDH];
  __shared__ __align__(16) _Float16 s_h1[32][LDH];
  __shared__ __align__(16) _Float16 s_h2[32][LDH];
  __shared__ int s_i[32], s_j[32];
  const int p0 = blockIdx.x * 32;
  const int tid = threadIdx.x, lane = tid & 31, wave = tid >> 5;
  if (tid < 32) { s_i[tid] = gnn[2 * (p0 + tid)]; s_j[tid] = gnn[2 * (p0 + tid) + 1]; }
  __syncthreads();
  {
    const int m = tid >> 3, q = tid & 7;
    const int c = q * 32;
    const _Float16* pi = state_h + (size_t)s_i[m] * HD + c;
    const _Float16* pj = state_h + (size_t)s_j[m] * HD + c;
    #pragma unroll
    for (int j = 0; j < 4; ++j)
      *(v8h*)&s_diff[m][c + 8 * j] = *(const v8h*)(pi + 8 * j) - *(const v8h*)(pj + 8 * j);
  }
  __syncthreads();

  for (int pass = 0; pass < 2; ++pass) {
    const _Float16* w1 = pass ? aw1 : tw1;  const float* bb1 = pass ? ab1 : tb1;
    const _Float16* w2 = pass ? aw2 : tw2;  const float* bb2 = pass ? ab2 : tb2;
    const _Float16* w3 = pass ? aw3 : tw3;  const float* bb3 = pass ? ab3 : tb3;
    float* outp = pass ? out1 : out0;

    // h1 = relu(diff @ w1^T + b1)
    #pragma unroll
    for (int j = 0; j < 2; ++j) {
      const int n0 = (wave * 2 + j) * 16;
      v8f acc0 = {}, acc1 = {};
      #pragma unroll
      for (int kt = 0; kt < HD / 32; ++kt) {
        v16h b = frag_b_glb(w1, HD, n0, kt, lane);
        acc0 = wmma16(frag_a_lds(&s_diff[0][0],  LDH, kt, lane), b, acc0);
        acc1 = wmma16(frag_a_lds(&s_diff[16][0], LDH, kt, lane), b, acc1);
      }
      const int n = n0 + (lane & 15);
      const float bias = bb1[n];
      const int mb = (lane & 16) ? 8 : 0;
      #pragma unroll
      for (int r = 0; r < 8; ++r) {
        float v0 = acc0[r] + bias, v1 = acc1[r] + bias;
        s_h1[mb + r][n]      = (_Float16)(v0 > 0.0f ? v0 : 0.0f);
        s_h1[16 + mb + r][n] = (_Float16)(v1 > 0.0f ? v1 : 0.0f);
      }
    }
    __syncthreads();

    // h2 = relu(h1 @ w2^T + b2)
    #pragma unroll
    for (int j = 0; j < 2; ++j) {
      const int n0 = (wave * 2 + j) * 16;
      v8f acc0 = {}, acc1 = {};
      #pragma unroll
      for (int kt = 0; kt < HD / 32; ++kt) {
        v16h b = frag_b_glb(w2, HD, n0, kt, lane);
        acc0 = wmma16(frag_a_lds(&s_h1[0][0],  LDH, kt, lane), b, acc0);
        acc1 = wmma16(frag_a_lds(&s_h1[16][0], LDH, kt, lane), b, acc1);
      }
      const int n = n0 + (lane & 15);
      const float bias = bb2[n];
      const int mb = (lane & 16) ? 8 : 0;
      #pragma unroll
      for (int r = 0; r < 8; ++r) {
        float v0 = acc0[r] + bias, v1 = acc1[r] + bias;
        s_h2[mb + r][n]      = (_Float16)(v0 > 0.0f ? v0 : 0.0f);
        s_h2[16 + mb + r][n] = (_Float16)(v1 > 0.0f ? v1 : 0.0f);
      }
    }
    __syncthreads();

    // out = h2 @ w3p^T + b3 (w3 zero-padded to 32 rows); waves 0..1 cover 32 cols
    if (wave < 2) {
      const int n0 = wave * 16;
      v8f acc0 = {}, acc1 = {};
      #pragma unroll
      for (int kt = 0; kt < HD / 32; ++kt) {
        v16h b = frag_b_glb(w3, HD, n0, kt, lane);
        acc0 = wmma16(frag_a_lds(&s_h2[0][0],  LDH, kt, lane), b, acc0);
        acc1 = wmma16(frag_a_lds(&s_h2[16][0], LDH, kt, lane), b, acc1);
      }
      const int n = n0 + (lane & 15);
      if (n < KM) {
        const float bias = bb3[n];
        const int mb = (lane & 16) ? 8 : 0;
        #pragma unroll
        for (int r = 0; r < 8; ++r) {
          outp[(size_t)(p0 + mb + r) * KM + n]      = acc0[r] + bias;
          outp[(size_t)(p0 + 16 + mb + r) * KM + n] = acc1[r] + bias;
        }
      }
    }
    __syncthreads();
  }
}

// ---------------- host-side orchestration ----------------
extern "C" void kernel_launch(void* const* d_in, const int* in_sizes, int n_in,
                              void* d_out, int out_size, void* d_ws, size_t ws_size,
                              hipStream_t stream) {
  const int* user_idx      = (const int*)  d_in[0];
  const int* node_idx_feat = (const int*)  d_in[1];
  const float* pos_idx     = (const float*)d_in[2];
  const int* edges         = (const int*)  d_in[3];
  const int* node_idx_gnn  = (const int*)  d_in[4];
  const int* att_idx       = (const int*)  d_in[5];
  const float* att_noise   = (const float*)d_in[6];
  const float* users_feat  = (const float*)d_in[7];
  const float* items_feat  = (const float*)d_in[8];
  const float* msg_w1 = (const float*)d_in[9];   const float* msg_b1 = (const float*)d_in[10];
  const float* msg_w2 = (const float*)d_in[11];  const float* msg_b2 = (const float*)d_in[12];
  const float* att_w1 = (const float*)d_in[13];  const float* att_b1 = (const float*)d_in[14];
  const float* att_w2 = (const float*)d_in[15];  const float* att_b2 = (const float*)d_in[16];
  const float* gru_wih = (const float*)d_in[17]; const float* gru_bih = (const float*)d_in[18];
  const float* gru_whh = (const float*)d_in[19]; const float* gru_bhh = (const float*)d_in[20];
  const float* tw1 = (const float*)d_in[21]; const float* tb1 = (const float*)d_in[22];
  const float* tw2 = (const float*)d_in[23]; const float* tb2 = (const float*)d_in[24];
  const float* tw3 = (const float*)d_in[25]; const float* tb3 = (const float*)d_in[26];
  const float* aw1 = (const float*)d_in[27]; const float* ab1 = (const float*)d_in[28];
  const float* aw2 = (const float*)d_in[29]; const float* ab2 = (const float*)d_in[30];
  const float* aw3 = (const float*)d_in[31]; const float* ab3 = (const float*)d_in[32];

  const int N = in_sizes[0];
  const int E = in_sizes[3] / 2;
  const int P = in_sizes[4] / 2;

  // workspace carve-up
  char* ws = (char*)d_ws;
  size_t off = 0;
  auto carve = [&](size_t bytes) -> void* {
    off = (off + 255) & ~(size_t)255;
    void* p = ws + off;
    off += bytes;
    return p;
  };
  float*     state   = (float*)    carve((size_t)N * HD * 4);
  _Float16*  state_h = (_Float16*) carve((size_t)N * HD * 2);
  float*     agg     = (float*)    carve((size_t)N * HD * 4);
  float*     sums    = (float*)    carve(256);
  _Float16* msg_w1h = (_Float16*)carve((size_t)LN * HD * (HD + EFD) * 2);
  _Float16* msg_w2h = (_Float16*)carve((size_t)LN * HD * HD * 2);
  _Float16* att_w1h = (_Float16*)carve((size_t)LN * AHD * (HD + EFD) * 2);
  _Float16* att_w2h = (_Float16*)carve((size_t)LN * HD * AHD * 2);
  _Float16* wihh    = (_Float16*)carve((size_t)LN * 3 * HD * HD * 2);
  _Float16* whhh    = (_Float16*)carve((size_t)LN * 3 * HD * HD * 2);
  _Float16* tw1h = (_Float16*)carve((size_t)HD * HD * 2);
  _Float16* tw2h = (_Float16*)carve((size_t)HD * HD * 2);
  _Float16* tw3p = (_Float16*)carve((size_t)KMP * HD * 2);
  _Float16* aw1h = (_Float16*)carve((size_t)HD * HD * 2);
  _Float16* aw2h = (_Float16*)carve((size_t)HD * HD * 2);
  _Float16* aw3p = (_Float16*)carve((size_t)KMP * HD * 2);
  (void)ws_size; (void)n_in; (void)out_size;

  float* out_theta = (float*)d_out;
  float* out_alpha = (float*)d_out + (size_t)P * KM;
  float* out_bpr   = (float*)d_out + 2 * (size_t)P * KM;

  auto cvt = [&](const float* s, _Float16* d, size_t n) {
    f32_to_f16_kernel<<<(unsigned)((n + 255) / 256), 256, 0, stream>>>(s, d, n);
  };

  // weight conversion (f32 -> f16, keep [out,in] row-major for contiguous B fragments)
  cvt(msg_w1, msg_w1h, (size_t)LN * HD * (HD + EFD));
  cvt(msg_w2, msg_w2h, (size_t)LN * HD * HD);
  cvt(att_w1, att_w1h, (size_t)LN * AHD * (HD + EFD));
  cvt(att_w2, att_w2h, (size_t)LN * HD * AHD);
  cvt(gru_wih, wihh,   (size_t)LN * 3 * HD * HD);
  cvt(gru_whh, whhh,   (size_t)LN * 3 * HD * HD);
  cvt(tw1, tw1h, (size_t)HD * HD);
  cvt(tw2, tw2h, (size_t)HD * HD);
  cvt(aw1, aw1h, (size_t)HD * HD);
  cvt(aw2, aw2h, (size_t)HD * HD);
  pad_w3_kernel<<<(KMP * HD + 255) / 256, 256, 0, stream>>>(tw3, tw3p);
  pad_w3_kernel<<<(KMP * HD + 255) / 256, 256, 0, stream>>>(aw3, aw3p);

  // BPR: sum(pos_idx) -> per-node loss; also write initial state = items_emb
  zero_f32_kernel<<<1, 256, 0, stream>>>(sums, 1);
  sum_pos_kernel<<<128, 256, 0, stream>>>(pos_idx, sums, N);
  node_init_kernel<<<(N + 255) / 256, 256, 0, stream>>>(
      user_idx, node_idx_feat, pos_idx, users_feat, items_feat, sums, state, out_bpr, N);

  const size_t NH = (size_t)N * HD;
  for (int li = 0; li < LN; ++li) {
    prep_state_kernel<<<(unsigned)((NH + 255) / 256), 256, 0, stream>>>(
        state, state_h, agg, /*relu=*/li > 0, NH);
    edge_msg_kernel<<<E / 32, 256, 0, stream>>>(
        state_h, edges, att_idx, att_noise,
        msg_w1h + (size_t)li * HD * (HD + EFD), msg_b1 + (size_t)li * HD,
        msg_w2h + (size_t)li * HD * HD,         msg_b2 + (size_t)li * HD,
        att_w1h + (size_t)li * AHD * (HD + EFD), att_b1 + (size_t)li * AHD,
        att_w2h + (size_t)li * HD * AHD,         att_b2 + (size_t)li * HD,
        agg);
    gru_kernel<<<(N + 31) / 32, 256, 0, stream>>>(
        agg, state_h,
        wihh + (size_t)li * 3 * HD * HD, whhh + (size_t)li * 3 * HD * HD,
        gru_bih + (size_t)li * 3 * HD,   gru_bhh + (size_t)li * 3 * HD,
        state, N);
  }

  // final state -> f16 (no relu), then output heads
  prep_state_kernel<<<(unsigned)((NH + 255) / 256), 256, 0, stream>>>(
      state, state_h, agg, /*relu=*/0, NH);
  head_kernel<<<P / 32, 256, 0, stream>>>(
      state_h, node_idx_gnn,
      tw1h, tb1, tw2h, tb2, tw3p, tb3,
      aw1h, ab1, aw2h, ab2, aw3p, ab3,
      out_theta, out_alpha);
}
